// FWDense_24223615549829
// MI455X (gfx1250) — matrix operand order repaired and checked by reading, not compiled, not run on hardware
//
#include <hip/hip_runtime.h>

// ---------------------------------------------------------------------------
// FWDense for MI455X (gfx1250, wave32, WMMA bf16).
// out[b,l,e] = x@W + b - step * ( q@cumKV(exclusive) + tril(q k^T,-1)@v )
//
// Multi-pass decomposition sized for MI455X: the per-chunk KV outer products
// and their exclusive cumsum are materialized as bf16 in d_ws (67MB -> lives
// in the 192MB global L2), unlocking 512-2048 concurrent workgroups instead
// of the 16 a fused serial-cumsum kernel allows. All WMMA operand gathers are
// lane-contiguous (ds_load_b128 / global_load_b128 pairs).
// Falls back to the fused single-kernel version if ws_size is too small.
// ---------------------------------------------------------------------------

typedef __bf16 bf16;
typedef __attribute__((ext_vector_type(16))) __bf16 v16bf;
typedef __attribute__((ext_vector_type(8)))  float  v8f;

#define NB 4
#define NL 4096
#define ND 512
#define NE 512
#define NG 32
#define CH 128          // chunk length (L / G)
#define ET 128          // e-tile
#define NTHREADS 256    // 8 waves

// ---- workspace layout (bf16) ----
// kvT: [b][g][e=512][s=512]  per-chunk KV, transposed, later exclusive-cumsum'd
// Wt : [e=512][s=512]        W transposed
#define KV_ELEMS   ((size_t)NB * NG * 512 * 512)
#define WT_ELEMS   ((size_t)512 * 512)
#define WS_NEEDED  ((KV_ELEMS + WT_ELEMS) * 2)

// --- wave32 WMMA operand k order used here: k = 16*(lane/16) + e, A and B alike ----------
__device__ __forceinline__ int a_k_of(int e, int lane) {
  const int j = e >> 1, p = e & 1;
  const int half = (lane < 16) ? 0 : 16;
  return 2 * j + half + p;
}

// A fragment from row-major [M][K] tile (contiguous dword pairs per lane)
__device__ __forceinline__ v16bf load_a_rm(const bf16* t, int ld, int m0, int k0, int lane) {
  v16bf r;
  const int m = m0 + (lane & 15);
  #pragma unroll
  for (int e = 0; e < 16; ++e) r[e] = t[m * ld + k0 + a_k_of(e, lane)];
  return r;
}

// B fragment; tile stores B^T row-major (t[n][k]) -> contiguous 32B per lane.
__device__ __forceinline__ v16bf load_b_tr(const bf16* t, int ld, int n0, int k0, int lane) {
  v16bf r;
  const int n  = n0 + (lane & 15);
  const int kb = (lane < 16) ? 0 : 16;
  #pragma unroll
  for (int e = 0; e < 16; ++e) {
    const int j = e >> 1, p = e & 1;
    r[e] = t[n * ld + k0 + kb + 2 * j + p];
  }
  return r;
}

// Same gather from a GLOBAL bf16 B^T tile (streams from L2; single-use data)
__device__ __forceinline__ v16bf load_b_tr_g(const bf16* __restrict__ t, int ld,
                                             int n0, int k0, int lane) {
  v16bf r;
  const int n  = n0 + (lane & 15);
  const int kb = (lane < 16) ? 0 : 16;
  #pragma unroll
  for (int e = 0; e < 16; ++e) {
    const int j = e >> 1, p = e & 1;
    r[e] = t[(size_t)n * ld + k0 + kb + 2 * j + p];
  }
  return r;
}

__device__ __forceinline__ v8f wmma_bf16(v16bf a, v16bf b, v8f c) {
  return __builtin_amdgcn_wmma_f32_16x16x32_bf16(false, a, false, b, (short)0, c, false, false);
}

// Stage 128x128 fp32 tile -> bf16 LDS row-major; float4 global reads
__device__ __forceinline__ void stage128(bf16* dst, const float* __restrict__ src,
                                         int ld, float scale, int tid) {
  #pragma unroll 4
  for (int i = 0; i < 16; ++i) {
    const int e = (i * NTHREADS + tid) * 4;
    const int r = e >> 7, c = e & 127;
    const float4 f = *(const float4*)(src + r * ld + c);
    dst[e + 0] = (bf16)(f.x * scale);
    dst[e + 1] = (bf16)(f.y * scale);
    dst[e + 2] = (bf16)(f.z * scale);
    dst[e + 3] = (bf16)(f.w * scale);
  }
}

// Stage 128x128 fp32 tile -> bf16 LDS TRANSPOSED: dst[c][r], row length ldd
__device__ __forceinline__ void stage128_t(bf16* dst, int ldd, const float* __restrict__ src,
                                           int ld, float scale, int tid) {
  #pragma unroll 4
  for (int i = 0; i < 16; ++i) {
    const int e = (i * NTHREADS + tid) * 4;
    const int r = e >> 7, c = e & 127;
    const float4 f = *(const float4*)(src + r * ld + c);
    dst[(c + 0) * ldd + r] = (bf16)(f.x * scale);
    dst[(c + 1) * ldd + r] = (bf16)(f.y * scale);
    dst[(c + 2) * ldd + r] = (bf16)(f.z * scale);
    dst[(c + 3) * ldd + r] = (bf16)(f.w * scale);
  }
}

// ===========================================================================
// Pass 0: W[s][e] -> Wt[e][s] bf16
// ===========================================================================
__global__ __launch_bounds__(NTHREADS)
void k_wt(const float* __restrict__ Wm, bf16* __restrict__ Wt) {
  const int f = blockIdx.x * NTHREADS + threadIdx.x;   // 512*512 threads
  const int e = f & 511, s = f >> 9;
  Wt[(size_t)e * 512 + s] = (bf16)Wm[(size_t)s * 512 + e];
}

// ===========================================================================
// Pass 1: kvT[b][g][e][s] = sum_c (-step * v[c,e]) * k[c,s]   (per chunk g)
// grid = b(4) x g(32) x et(4) x st(4) = 2048 WGs
// ===========================================================================
#define KV_SMEM (2 * 128 * 128 * 2)
__global__ __launch_bounds__(NTHREADS)
void k_kv(const float* __restrict__ swx, const float* __restrict__ grad,
          const float* __restrict__ step_ptr, bf16* __restrict__ kvT) {
  extern __shared__ char smem_raw[];
  bf16* sVt = (bf16*)smem_raw;                 // [e=128][c=128]  (= v^T, scaled)
  bf16* sKt = (bf16*)smem_raw + 128 * 128;     // [s=128][c=128]  (= k^T)

  const int tid  = threadIdx.x;
  const int lane = tid & 31;
  const int m0   = (tid >> 5) * 16;
  const int half = (lane < 16) ? 0 : 8;
  const int nl   = lane & 15;

  const int wg = blockIdx.x;
  const int st = wg & 3;
  const int et = (wg >> 2) & 3;
  const int g  = (wg >> 4) & 31;
  const int b  = wg >> 9;

  const float negstep = -step_ptr[0];
  const float* kg = swx  + ((size_t)b * NL + g * CH) * ND;
  const float* vg = grad + ((size_t)b * NL + g * CH) * NE + et * 128;

  stage128_t(sVt, 128, vg, NE, negstep, tid);          // A = v^T (pre-scaled)
  stage128_t(sKt, 128, kg + st * 128, ND, 1.0f, tid);  // B^T = k^T
  __syncthreads();

  const v8f vzero = {0.f, 0.f, 0.f, 0.f, 0.f, 0.f, 0.f, 0.f};
  v8f cc[8];
  #pragma unroll
  for (int n = 0; n < 8; ++n) cc[n] = vzero;

  #pragma unroll 1
  for (int kk = 0; kk < 4; ++kk) {
    const v16bf a = load_a_rm(sVt, 128, m0, kk * 32, lane);
    #pragma unroll
    for (int n = 0; n < 8; ++n)
      cc[n] = wmma_bf16(a, load_b_tr(sKt, 128, n * 16, kk * 32, lane), cc[n]);
  }

  bf16* og = kvT + (((size_t)b * NG + g) * 512 + et * 128) * 512 + st * 128;
  #pragma unroll
  for (int n = 0; n < 8; ++n)
    #pragma unroll
    for (int j = 0; j < 8; ++j)
      og[(size_t)(m0 + j + half) * 512 + n * 16 + nl] = (bf16)cc[n][j];
}

// ===========================================================================
// Pass 2: in-place EXCLUSIVE cumsum over g, per (b,e,s) element, fp32 accum
// ===========================================================================
__global__ __launch_bounds__(NTHREADS)
void k_cumsum(bf16* __restrict__ kvT) {
  const size_t f = (size_t)blockIdx.x * NTHREADS + threadIdx.x;  // b*512*512
  const int s = (int)(f & 511);
  const int e = (int)((f >> 9) & 511);
  const int b = (int)(f >> 18);
  bf16* p = kvT + (((size_t)b * NG) * 512 + e) * 512 + s;
  const size_t gstride = (size_t)512 * 512;
  float acc = 0.f;
  #pragma unroll 1
  for (int g = 0; g < NG; ++g) {
    const float v = (float)p[g * gstride];
    p[g * gstride] = (bf16)acc;      // exclusive prefix
    acc += v;
  }
}

// ===========================================================================
// Pass 3: out = x@W + b + q@cumKV + tril(q k^T,-1)@(-step v)
// grid = b(4) x g(32) x et(4) = 512 WGs
// ===========================================================================
#define MAIN_SMEM (4 * 128 * 128 * 2 + ET * 4)
__global__ __launch_bounds__(NTHREADS)
void k_main(const float* __restrict__ x, const float* __restrict__ swx,
            const float* __restrict__ grad, const float* __restrict__ bias,
            const float* __restrict__ step_ptr,
            const bf16* __restrict__ kvT, const bf16* __restrict__ Wt,
            float* __restrict__ out) {
  extern __shared__ char smem_raw[];
  bf16*  sQ  = (bf16*)smem_raw;                  // [c=128][s=128] row-major
  bf16*  sK  = (bf16*)smem_raw + 128 * 128;      // [c=128][s=128] row-major
  bf16*  sVt = (bf16*)smem_raw + 2 * 128 * 128;  // [e=128][c=128], scaled -step
  bf16*  sQK = (bf16*)smem_raw + 3 * 128 * 128;  // [query][key]
  float* sB  = (float*)(smem_raw + 4 * 128 * 128 * 2);

  const int tid  = threadIdx.x;
  const int lane = tid & 31;
  const int m0   = (tid >> 5) * 16;
  const int half = (lane < 16) ? 0 : 8;
  const int nl   = lane & 15;

  const int wg = blockIdx.x;
  const int et = wg & 3;
  const int g  = (wg >> 2) & 31;
  const int b  = wg >> 7;
  const int e0 = et * ET;

  const float negstep = -step_ptr[0];
  const float* xg = x    + ((size_t)b * NL + g * CH) * ND;
  const float* kg = swx  + ((size_t)b * NL + g * CH) * ND;
  const float* vg = grad + ((size_t)b * NL + g * CH) * NE + e0;
  const bf16* kvg = kvT + (((size_t)b * NG + g) * 512 + e0) * 512;  // [e][s] tile
  const bf16* wtg = Wt + (size_t)e0 * 512;                          // [e][s] tile

  if (tid < ET) sB[tid] = bias[e0 + tid];
  stage128_t(sVt, 128, vg, NE, negstep, tid);

  const v8f vzero = {0.f, 0.f, 0.f, 0.f, 0.f, 0.f, 0.f, 0.f};
  v8f outAcc[8], qkAcc[8];
  #pragma unroll
  for (int n = 0; n < 8; ++n) { outAcc[n] = vzero; qkAcc[n] = vzero; }

  // ---- qk + glob + dense over the s dimension (4 sub-tiles of 128) ----
  #pragma unroll 1
  for (int ss = 0; ss < 4; ++ss) {
    __syncthreads();
    stage128(sQ, xg + ss * 128, ND, 1.0f, tid);
    stage128(sK, kg + ss * 128, ND, 1.0f, tid);
    __syncthreads();
    #pragma unroll 1
    for (int kk = 0; kk < 4; ++kk) {
      const int s0 = ss * 128 + kk * 32;
      const v16bf a = load_a_rm(sQ, 128, m0, kk * 32, lane);
      #pragma unroll
      for (int n = 0; n < 8; ++n) {
        qkAcc[n]  = wmma_bf16(a, load_b_tr(sK, 128, n * 16, kk * 32, lane), qkAcc[n]);
        outAcc[n] = wmma_bf16(a, load_b_tr_g(kvg, 512, n * 16, s0, lane), outAcc[n]); // glob (L2)
        outAcc[n] = wmma_bf16(a, load_b_tr_g(wtg, 512, n * 16, s0, lane), outAcc[n]); // dense (L2)
      }
    }
  }
  __syncthreads();

  // ---- mask qk strictly-causal, spill bf16 ----
  #pragma unroll
  for (int n = 0; n < 8; ++n)
    #pragma unroll
    for (int j = 0; j < 8; ++j) {
      const int M = m0 + j + half, N = n * 16 + nl;
      sQK[M * 128 + N] = (bf16)((N < M) ? qkAcc[n][j] : 0.0f);
    }
  __syncthreads();

  // ---- local: out += qk @ (-step v) ----
  #pragma unroll 1
  for (int kk = 0; kk < 4; ++kk) {
    const v16bf a = load_a_rm(sQK, 128, m0, kk * 32, lane);
    #pragma unroll
    for (int n = 0; n < 8; ++n)
      outAcc[n] = wmma_bf16(a, load_b_tr(sVt, 128, n * 16, kk * 32, lane), outAcc[n]);
  }

  // ---- store (+bias) ----
  float* og = out + ((size_t)b * NL + g * CH + m0) * NE + e0;
  #pragma unroll
  for (int n = 0; n < 8; ++n)
    #pragma unroll
    for (int j = 0; j < 8; ++j)
      og[(size_t)(j + half) * NE + n * 16 + nl] = outAcc[n][j] + sB[n * 16 + nl];
}

// ===========================================================================
// Fallback: fused single-kernel (16 WGs, KV resident in LDS) if ws too small
// ===========================================================================
#define KV_OFF  0
#define Q_OFF   (512*128)
#define K_OFF   (Q_OFF  + 128*128)
#define W_OFF   (K_OFF  + 128*128)
#define V_OFF   (W_OFF  + 128*128)
#define QK_OFF  (V_OFF  + 128*128)
#define BF16_CNT (QK_OFF + 128*128)
#define FUSED_SMEM (BF16_CNT*2 + ET*4)

__global__ __launch_bounds__(NTHREADS)
void fwdense_fused(const float* __restrict__ x, const float* __restrict__ swx,
                   const float* __restrict__ grad, const float* __restrict__ Wm,
                   const float* __restrict__ bias, const float* __restrict__ step_ptr,
                   float* __restrict__ out) {
  extern __shared__ char smem_raw[];
  bf16*  sKVt = (bf16*)smem_raw + KV_OFF;
  bf16*  sQ   = (bf16*)smem_raw + Q_OFF;
  bf16*  sK   = (bf16*)smem_raw + K_OFF;
  bf16*  sWt  = (bf16*)smem_raw + W_OFF;
  bf16*  sVt  = (bf16*)smem_raw + V_OFF;
  bf16*  sQK  = (bf16*)smem_raw + QK_OFF;
  float* sB   = (float*)(smem_raw + BF16_CNT * 2);

  const int tid  = threadIdx.x;
  const int lane = tid & 31;
  const int m0   = (tid >> 5) * 16;
  const int half = (lane < 16) ? 0 : 8;
  const int nl   = lane & 15;
  const int b    = blockIdx.x >> 2;
  const int e0   = (blockIdx.x & 3) * ET;
  const float negstep = -step_ptr[0];

  {
    uint4* p = (uint4*)sKVt;
    const uint4 z = {0u, 0u, 0u, 0u};
    #pragma unroll
    for (int i = 0; i < 32; ++i) p[i * NTHREADS + tid] = z;
  }
  if (tid < ET) sB[tid] = bias[e0 + tid];
  __syncthreads();

  const v8f vzero = {0.f, 0.f, 0.f, 0.f, 0.f, 0.f, 0.f, 0.f};
  #pragma unroll 1
  for (int g = 0; g < NG; ++g) {
    const float* xg = x    + ((size_t)b * NL + g * CH) * ND;
    const float* kg = swx  + ((size_t)b * NL + g * CH) * ND;
    const float* vg = grad + ((size_t)b * NL + g * CH) * NE + e0;
    v8f outAcc[8], qkAcc[8];
    #pragma unroll
    for (int n = 0; n < 8; ++n) { outAcc[n] = vzero; qkAcc[n] = vzero; }
    stage128_t(sVt, 128, vg, NE, negstep, tid);
    #pragma unroll 1
    for (int ss = 0; ss < 4; ++ss) {
      __syncthreads();
      stage128  (sQ,       xg + ss * 128, ND, 1.0f, tid);
      stage128  (sK,       kg + ss * 128, ND, 1.0f, tid);
      stage128_t(sWt, 128, Wm + (size_t)(ss * 128) * NE + e0, NE, 1.0f, tid);
      __syncthreads();
      #pragma unroll 1
      for (int kk = 0; kk < 4; ++kk) {
        const v16bf a = load_a_rm(sQ, 128, m0, kk * 32, lane);
        #pragma unroll
        for (int n = 0; n < 8; ++n) {
          qkAcc[n]  = wmma_bf16(a, load_b_tr(sK,  128, n * 16, kk * 32, lane), qkAcc[n]);
          outAcc[n] = wmma_bf16(a, load_b_tr(sKVt, 512, n * 16, ss * 128 + kk * 32, lane), outAcc[n]);
          outAcc[n] = wmma_bf16(a, load_b_tr(sWt, 128, n * 16, kk * 32, lane), outAcc[n]);
        }
      }
    }
    __syncthreads();
    #pragma unroll
    for (int n = 0; n < 8; ++n)
      #pragma unroll
      for (int j = 0; j < 8; ++j) {
        const int M = m0 + j + half, N = n * 16 + nl;
        sQK[M * 128 + N] = (bf16)((N < M) ? qkAcc[n][j] : 0.0f);
      }
    __syncthreads();
    #pragma unroll 1
    for (int kk = 0; kk < 4; ++kk) {
      const v16bf a = load_a_rm(sQK, 128, m0, kk * 32, lane);
      #pragma unroll
      for (int n = 0; n < 8; ++n)
        outAcc[n] = wmma_bf16(a, load_b_tr(sVt, 128, n * 16, kk * 32, lane), outAcc[n]);
    }
    {
      float* og = out + ((size_t)b * NL + g * CH + m0) * NE + e0;
      #pragma unroll
      for (int n = 0; n < 8; ++n)
        #pragma unroll
        for (int j = 0; j < 8; ++j)
          og[(size_t)(j + half) * NE + n * 16 + nl] = outAcc[n][j] + sB[n * 16 + nl];
    }
    #pragma unroll 1
    for (int ss = 0; ss < 4; ++ss) {
      __syncthreads();
      stage128_t(sK, 128, kg + ss * 128, ND, 1.0f, tid);
      __syncthreads();
      const int s0 = ss * 128 + m0;
      v8f cc[8];
      #pragma unroll
      for (int n = 0; n < 8; ++n)
        #pragma unroll
        for (int j = 0; j < 8; ++j)
          cc[n][j] = (float)sKVt[(n * 16 + nl) * 512 + s0 + half + j];
      #pragma unroll 1
      for (int kk = 0; kk < 4; ++kk) {
        const v16bf a = load_a_rm(sK, 128, m0, kk * 32, lane);
        #pragma unroll
        for (int n = 0; n < 8; ++n)
          cc[n] = wmma_bf16(a, load_b_tr(sVt, 128, n * 16, kk * 32, lane), cc[n]);
      }
      #pragma unroll
      for (int n = 0; n < 8; ++n)
        #pragma unroll
        for (int j = 0; j < 8; ++j)
          sKVt[(n * 16 + nl) * 512 + s0 + half + j] = (bf16)cc[n][j];
    }
    __syncthreads();
  }
}

// ===========================================================================
extern "C" void kernel_launch(void* const* d_in, const int* in_sizes, int n_in,
                              void* d_out, int out_size, void* d_ws, size_t ws_size,
                              hipStream_t stream) {
  (void)in_sizes; (void)n_in; (void)out_size;
  const float* x    = (const float*)d_in[0];
  const float* swx  = (const float*)d_in[1];
  const float* grad = (const float*)d_in[2];
  const float* Wm   = (const float*)d_in[3];
  const float* bias = (const float*)d_in[4];
  const float* step = (const float*)d_in[5];
  float* out = (float*)d_out;

  if (ws_size >= WS_NEEDED) {
    bf16* kvT = (bf16*)d_ws;
    bf16* Wt  = kvT + KV_ELEMS;

    (void)hipFuncSetAttribute((const void*)k_kv,
                              hipFuncAttributeMaxDynamicSharedMemorySize, (int)KV_SMEM);
    (void)hipFuncSetAttribute((const void*)k_main,
                              hipFuncAttributeMaxDynamicSharedMemorySize, (int)MAIN_SMEM);

    k_wt<<<dim3(512 * 512 / NTHREADS), dim3(NTHREADS), 0, stream>>>(Wm, Wt);
    k_kv<<<dim3(NB * NG * 4 * 4), dim3(NTHREADS), KV_SMEM, stream>>>(swx, grad, step, kvT);
    k_cumsum<<<dim3(NB * 512 * 512 / NTHREADS), dim3(NTHREADS), 0, stream>>>(kvT);
    k_main<<<dim3(NB * NG * 4), dim3(NTHREADS), MAIN_SMEM, stream>>>(
        x, swx, grad, bias, step, kvT, Wt, out);
  } else {
    (void)hipFuncSetAttribute((const void*)fwdense_fused,
                              hipFuncAttributeMaxDynamicSharedMemorySize, (int)FUSED_SMEM);
    fwdense_fused<<<dim3(NB * (NE / ET)), dim3(NTHREADS), FUSED_SMEM, stream>>>(
        x, swx, grad, Wm, bias, step, out);
  }
}
